// LatentUpdate_32933809226364
// MI455X (gfx1250) — compile-verified, matrix operand order checked by reading, not run on hardware
//
#include <hip/hip_runtime.h>
#include <hip/hip_bf16.h>
#include <math.h>

typedef __attribute__((ext_vector_type(16))) _Float16 v16h;
typedef __attribute__((ext_vector_type(8)))  float    v8f;
typedef __attribute__((ext_vector_type(4)))  unsigned int v4u;
typedef __attribute__((ext_vector_type(8)))  int      v8i;
typedef __attribute__((ext_vector_type(4)))  int      v4i;

#define KNN 30
#define KPAD 96
// packed weight block layout (in halves): contiguous 16KB for one TDM descriptor
#define WOFF_A1C 0        // 128x32
#define WOFF_WE1 4096     // 32x32
#define WOFF_WE2 5120     // 32x32
#define WOFF_WG  6144     // 64x32
#define WPACK_HALVES 8192 // 16384 bytes

static __device__ __forceinline__ float sigmoidf_(float x) { return 1.f / (1.f + expf(-x)); }
static __device__ __forceinline__ float siluf_(float x)    { return x / (1.f + expf(-x)); }

// Load a 16x32 f16 WMMA operand fragment from row-major storage (rows = M for A
// or N for pre-transposed B).  Per ISA 7.12.2: lane L holds row L%16; half h maps
// to K = (h/8)*16 + (L>=16 ? 8 : 0) + h%8  ->  two contiguous 16B segments.
static __device__ __forceinline__ v16h frag_ld(const _Float16* base, int row, int ld,
                                               int kc, int lane) {
  const _Float16* pr = base + (size_t)row * ld + kc + ((lane & 16) ? 8 : 0);
  v16h v;
#pragma unroll
  for (int i = 0; i < 8; ++i) v[i] = pr[i];
#pragma unroll
  for (int i = 0; i < 8; ++i) v[8 + i] = pr[16 + i];
  return v;
}

// Tensor Data Mover: DMA one contiguous 16KB block (2048 x 8B) global -> LDS.
// D# per ISA ch8: group0 = {count=1, lds_addr, global_addr, type=2};
// group1 = {data_size=8B, tensor_dim0 = tile_dim0 = 2048, dim0_stride = 2048}.
static __device__ __forceinline__ void tdm_copy_16k(const _Float16* gsrc, unsigned ldsAddr) {
  unsigned long long ga = (unsigned long long)gsrc;
  v4u g0;
  g0[0] = 1u;                                        // count=1 (user descriptor)
  g0[1] = ldsAddr;                                   // lds_addr (bytes)
  g0[2] = (unsigned)(ga & 0xffffffffu);              // global_addr[31:0]
  g0[3] = (unsigned)((ga >> 32) & 0x01ffffffu)       // global_addr[56:32]
        | (2u << 30);                                // type = 2 ("image")
  v8i g1 = {};
  g1[0] = (3 << 16);                                 // data_size = 3 (8 bytes)
  g1[1] = (int)(2048u << 16);                        // tensor_dim0[15:0] = 2048
  g1[2] = 0;                                         // tensor_dim0 hi / tensor_dim1 lo
  g1[3] = (int)(2048u << 16);                        // tile_dim0 = 2048
  g1[4] = 0;                                         // tile_dim1/2 unused (1D tile)
  g1[5] = 2048;                                      // tensor_dim0_stride
  g1[6] = 0; g1[7] = 0;
  v4i z4 = {};
#if __clang_major__ >= 23
  v8i z8 = {};
  __builtin_amdgcn_tensor_load_to_lds(g0, g1, z4, z4, z8, 0);
#else
  __builtin_amdgcn_tensor_load_to_lds(g0, g1, z4, z4, 0);
#endif
}

// ---------------- weight pre-pack: f32 row-major -> f16 N-major, K zero-padded ----
__global__ void pack_wt(const float* __restrict__ W, _Float16* __restrict__ Wt,
                        int ldw, int rowOff, int K, int Kpad, int Nout) {
  int idx = blockIdx.x * blockDim.x + threadIdx.x;
  int total = Nout * Kpad;
  if (idx >= total) return;
  int o = idx / Kpad, k2 = idx - o * Kpad;
  float v = (k2 < K) ? W[(size_t)(rowOff + k2) * ldw + o] : 0.f;
  Wt[idx] = (_Float16)v;
}

// ---------------- fused SO3 node embedding -> f16 padded [N*4, 96] ---------------
__global__ void node_feat(const float* __restrict__ rot, const float* __restrict__ rigid,
                          const float* __restrict__ side, const int* __restrict__ xmask,
                          const int* __restrict__ nmask, _Float16* __restrict__ X, int n) {
  int idx = blockIdx.x * blockDim.x + threadIdx.x;
  int total = n * 4 * KPAD;
  if (idx >= total) return;
  int c = idx % KPAD;
  int l = (idx / KPAD) & 3;
  int node = idx / (KPAD * 4);
  float v = 0.f;
  if (c < 32) {
    v = rigid[((size_t)node * 4 + l) * 32 + c];
  } else if (c < 64) {
    v = side[((size_t)node * 4 + l) * 32 + (c - 32)];
  } else if (c < 67) {
    int a = c - 64;
    if (l == 0) {
      v = (a == 2) ? ((nmask[node] != 0 && xmask[node] == 0) ? 1.f : 0.f) : 0.f;
    } else {
      const float atoms[3][3] = {{-0.525f, 1.363f, 0.f},
                                 {1.526f, 0.f, 0.f},
                                 {-0.529f, -0.774f, -1.205f}};
      int i = l - 1;
      v = rot[(size_t)node * 9 + i * 3 + 0] * atoms[a][0]
        + rot[(size_t)node * 9 + i * 3 + 1] * atoms[a][1]
        + rot[(size_t)node * 9 + i * 3 + 2] * atoms[a][2];
    }
  }
  X[idx] = (_Float16)v;
}

// ---------------- KNN: one wave per node, iterative min with (d2,idx) keys -------
__global__ void knn_kernel(const float* __restrict__ trans, const int* __restrict__ batch,
                           int* __restrict__ srcIdx, int* __restrict__ validE,
                           int n, int K) {
  int i = blockIdx.x;
  int lane = threadIdx.x;
  int b = batch[i];
  int lo = 0, hi = n;                       // segment start (batch is sorted)
  while (lo < hi) { int mid = (lo + hi) >> 1; if (batch[mid] < b) lo = mid + 1; else hi = mid; }
  int segs = lo;
  lo = 0; hi = n;                           // segment end
  while (lo < hi) { int mid = (lo + hi) >> 1; if (batch[mid] <= b) lo = mid + 1; else hi = mid; }
  int sege = lo;
  float xi = trans[3 * i], yi = trans[3 * i + 1], zi = trans[3 * i + 2];
  float prevD = -1.f; int prevJ = -1;
  for (int t = 0; t < K; ++t) {
    float bestD = 3.0e38f; int bestJ = 0x7fffffff;
    for (int j = segs + lane; j < sege; j += 32) {
      if (j == i) continue;
      float dx = trans[3 * j] - xi, dy = trans[3 * j + 1] - yi, dz = trans[3 * j + 2] - zi;
      float d2 = dx * dx + dy * dy + dz * dz;
      bool gt = (d2 > prevD) || (d2 == prevD && j > prevJ);
      if (gt && (d2 < bestD || (d2 == bestD && j < bestJ))) { bestD = d2; bestJ = j; }
    }
    for (int m = 16; m >= 1; m >>= 1) {
      float od = __shfl_xor(bestD, m, 32);
      int   oj = __shfl_xor(bestJ, m, 32);
      if (od < bestD || (od == bestD && oj < bestJ)) { bestD = od; bestJ = oj; }
    }
    int found = (bestJ != 0x7fffffff) && (bestD < 1e17f);
    if (lane == 0) {
      srcIdx[(size_t)i * K + t] = found ? bestJ : i;
      validE[(size_t)i * K + t] = found;
    }
    if (found) { prevD = bestD; prevJ = bestJ; }
    else       { prevD = 3.0e38f; prevJ = 0x7fffffff; }
  }
}

// ---------------- generic 16x16 WMMA GEMM: C[row,n] = A[row,:] * Bt[n,:]^T --------
__global__ void gemm16(const _Float16* __restrict__ A, int lda,
                       const _Float16* __restrict__ Bt, int ldb, int kTiles,
                       float* __restrict__ C, int ldc) {
  int lane = threadIdx.x;
  int row0 = blockIdx.x * 16;
  int n0 = blockIdx.y * 16;
  v8f acc = {};
  for (int kt = 0; kt < kTiles; ++kt) {
    v16h a = frag_ld(A, row0 + (lane & 15), lda, kt * 32, lane);
    v16h b = frag_ld(Bt, n0 + (lane & 15), ldb, kt * 32, lane);
    acc = __builtin_amdgcn_wmma_f32_16x16x32_f16(false, a, false, b, (short)0, acc,
                                                 false, false);
  }
  int mb = (lane & 16) ? 8 : 0, nn = (lane & 15);
#pragma unroll
  for (int r = 0; r < 8; ++r)
    C[(size_t)(row0 + mb + r) * ldc + n0 + nn] = acc[r];
}

// ---------------- fused per-edge pipeline: 16 edges per wave, WMMA MLP chain ------
// Weights (Wa1c|We1|We2|Wg, 16KB) staged into LDS by the Tensor Data Mover,
// overlapped with edge-scalar prep; s_wait_tensorcnt before first B-frag read.
__global__ void edge_kernel(const float* __restrict__ trans, const int* __restrict__ srcIdx,
                            const int* __restrict__ validE,
                            const _Float16* __restrict__ Wpack,
                            const float* __restrict__ Wa2, const float* __restrict__ p,
                            const float* __restrict__ q, float* __restrict__ logits,
                            _Float16* __restrict__ gateOut, int K, long Etot) {
  __shared__ float s_dist[16], s_dseq[16];
  __shared__ int s_src[16], s_dst[16], s_valid[16];
  __shared__ _Float16 tile[16 * 32];
  __shared__ _Float16 wlds[WPACK_HALVES];      // 16KB TDM-staged weights
  int lane = threadIdx.x;
  long e0 = (long)blockIdx.x * 16;

  // kick off the async weight DMA first (TENSORcnt), overlap with edge prep
  tdm_copy_16k(Wpack, (unsigned)(size_t)(&wlds[0]));

  if (lane < 16) {
    long ge = e0 + lane;
    if (ge < Etot) {
      int d = (int)(ge / K);
      int s = srcIdx[ge];
      __builtin_prefetch(p + (size_t)s * 128, 0, 1);   // global_prefetch_b8
      __builtin_prefetch(q + (size_t)d * 128, 0, 1);
      float dx = trans[3 * s] - trans[3 * d];
      float dy = trans[3 * s + 1] - trans[3 * d + 1];
      float dz = trans[3 * s + 2] - trans[3 * d + 2];
      float dist = sqrtf(dx * dx + dy * dy + dz * dz);
      int v = validE[ge] && (dist > 1e-3f) && (dist < 3.0e38f) && (dist == dist);
      s_dist[lane] = dist; s_dseq[lane] = (float)(s - d);
      s_src[lane] = s; s_dst[lane] = d; s_valid[lane] = v;
    } else {
      s_dist[lane] = 0.f; s_dseq[lane] = 0.f;
      s_src[lane] = 0; s_dst[lane] = 0; s_valid[lane] = 0;
    }
  }
  __syncthreads();
  // ---- edge scalar features: 16 gaussian RBF + 8 cos + 8 sin (lane = feature) ----
  {
    float mu = 0.f, fr = 0.f; int kind;
    const float c0 = 9.210340371976184f / 16.f;   // ln(10000)/16
    if (lane < 16)      { kind = 0; mu = lane * (20.f / 15.f); }
    else if (lane < 24) { kind = 1; fr = expf(-(float)(2 * (lane - 16)) * c0); }
    else                { kind = 2; fr = expf(-(float)(2 * (lane - 24)) * c0); }
    for (int e = 0; e < 16; ++e) {
      float v;
      if (kind == 0)      { float r = (s_dist[e] - mu) * (1.f / 1.25f); v = expf(-r * r); }
      else if (kind == 1) v = cosf(s_dseq[e] * fr);
      else                v = sinf(s_dseq[e] * fr);
      tile[e * 32 + lane] = (_Float16)v;
    }
  }
  __syncthreads();
  // weights must be resident in LDS from here on
  __builtin_amdgcn_s_wait_tensorcnt(0);
  asm volatile("" ::: "memory");   // LDS now holds TDM data; block reordering
  __syncthreads();

  int mb = (lane & 16) ? 8 : 0, nn = (lane & 15);
  v16h aF = frag_ld(tile, nn, 32, 0, lane);
  __syncthreads();
  { // h1 = silu(ef @ We1)   [16x32]
    v16h b0 = frag_ld(wlds + WOFF_WE1, nn, 32, 0, lane);
    v16h b1 = frag_ld(wlds + WOFF_WE1, 16 + nn, 32, 0, lane);
    v8f a0 = {}, a1 = {};
    a0 = __builtin_amdgcn_wmma_f32_16x16x32_f16(false, aF, false, b0, (short)0, a0, false, false);
    a1 = __builtin_amdgcn_wmma_f32_16x16x32_f16(false, aF, false, b1, (short)0, a1, false, false);
#pragma unroll
    for (int r = 0; r < 8; ++r) {
      tile[(mb + r) * 32 + nn]      = (_Float16)siluf_(a0[r]);
      tile[(mb + r) * 32 + 16 + nn] = (_Float16)siluf_(a1[r]);
    }
  }
  __syncthreads();
  aF = frag_ld(tile, nn, 32, 0, lane);
  __syncthreads();
  { // xe = h1 @ We2   [16x32]
    v16h b0 = frag_ld(wlds + WOFF_WE2, nn, 32, 0, lane);
    v16h b1 = frag_ld(wlds + WOFF_WE2, 16 + nn, 32, 0, lane);
    v8f a0 = {}, a1 = {};
    a0 = __builtin_amdgcn_wmma_f32_16x16x32_f16(false, aF, false, b0, (short)0, a0, false, false);
    a1 = __builtin_amdgcn_wmma_f32_16x16x32_f16(false, aF, false, b1, (short)0, a1, false, false);
#pragma unroll
    for (int r = 0; r < 8; ++r) {
      tile[(mb + r) * 32 + nn]      = (_Float16)a0[r];
      tile[(mb + r) * 32 + 16 + nn] = (_Float16)a1[r];
    }
  }
  __syncthreads();
  aF = frag_ld(tile, nn, 32, 0, lane);
  // ---- logits: alpha = silu(p[src]+q[dst]+xe@Wa1c), reduce with Wa2 per head ----
#pragma unroll 1
  for (int t = 0; t < 8; ++t) {          // n-tile t == head t (16 alpha channels)
    v16h b = frag_ld(wlds + WOFF_A1C, t * 16 + nn, 32, 0, lane);
    v8f acc = {};
    acc = __builtin_amdgcn_wmma_f32_16x16x32_f16(false, aF, false, b, (short)0, acc, false, false);
    float w2 = Wa2[t * 16 + nn];
#pragma unroll
    for (int r = 0; r < 8; ++r) {
      int e = mb + r;
      int col = t * 16 + nn;
      float val = acc[r] + p[(size_t)s_src[e] * 128 + col] + q[(size_t)s_dst[e] * 128 + col];
      float lg = siluf_(val) * w2;
      lg += __shfl_xor(lg, 8, 16);
      lg += __shfl_xor(lg, 4, 16);
      lg += __shfl_xor(lg, 2, 16);
      lg += __shfl_xor(lg, 1, 16);
      if (nn == 0 && (e0 + e) < Etot)
        logits[(e0 + e) * 8 + t] = s_valid[e] ? lg : -1e9f;
    }
  }
  // ---- gate = sigmoid(xe @ Wg) * valid  ->  f16 [E,64] -----------------------------
#pragma unroll 1
  for (int t = 0; t < 4; ++t) {
    v16h b = frag_ld(wlds + WOFF_WG, t * 16 + nn, 32, 0, lane);
    v8f acc = {};
    acc = __builtin_amdgcn_wmma_f32_16x16x32_f16(false, aF, false, b, (short)0, acc, false, false);
#pragma unroll
    for (int r = 0; r < 8; ++r) {
      int e = mb + r;
      if ((e0 + e) < Etot) {
        float g = s_valid[e] ? sigmoidf_(acc[r]) : 0.f;
        gateOut[(e0 + e) * 64 + t * 16 + nn] = (_Float16)g;
      }
    }
  }
}

// ---------------- per-node: softmax over k, gather-combine, Wo, scalar-gated Wf ----
__global__ void combine_kernel(const float* __restrict__ logits, const _Float16* __restrict__ gate,
                               const int* __restrict__ srcIdx, const float* __restrict__ y,
                               const float* __restrict__ Wo, const float* __restrict__ Wf,
                               const int* __restrict__ nmask, const float* __restrict__ side,
                               float* __restrict__ out) {
  __shared__ float attn[KNN * 8];
  __shared__ float ob[4 * 64];
  __shared__ float u0[32];
  int n = blockIdx.x;
  int lane = threadIdx.x;
  if (lane < 8) {
    float m = -3.4e38f;
    for (int k2 = 0; k2 < KNN; ++k2)
      m = fmaxf(m, logits[((size_t)n * KNN + k2) * 8 + lane]);
    float s = 0.f;
    for (int k2 = 0; k2 < KNN; ++k2) {
      float e = expf(logits[((size_t)n * KNN + k2) * 8 + lane] - m);
      attn[k2 * 8 + lane] = e; s += e;
    }
    float inv = 1.f / s;
    for (int k2 = 0; k2 < KNN; ++k2) attn[k2 * 8 + lane] *= inv;
  }
  __syncthreads();
  float acc[4][2] = {};
  for (int k2 = 0; k2 < KNN; ++k2) {
    size_t ge = (size_t)n * KNN + k2;
    int s = srcIdx[ge];
#pragma unroll
    for (int ci = 0; ci < 2; ++ci) {
      int c = lane + 32 * ci;
      float w = attn[k2 * 8 + (c >> 3)] * (float)gate[ge * 64 + c];
#pragma unroll
      for (int l2 = 0; l2 < 4; ++l2)
        acc[l2][ci] += w * y[((size_t)s * 4 + l2) * 64 + c];
    }
  }
#pragma unroll
  for (int ci = 0; ci < 2; ++ci)
#pragma unroll
    for (int l2 = 0; l2 < 4; ++l2) ob[l2 * 64 + lane + 32 * ci] = acc[l2][ci];
  __syncthreads();
  float u[4];
#pragma unroll
  for (int l2 = 0; l2 < 4; ++l2) {
    float s2 = 0.f;
    for (int c = 0; c < 64; ++c) s2 += ob[l2 * 64 + c] * Wo[c * 32 + lane];
    u[l2] = s2;
  }
  u0[lane] = u[0];
  __syncthreads();
  float g2 = 0.f;
  for (int c = 0; c < 32; ++c) g2 += u0[c] * Wf[c * 32 + lane];
  g2 = sigmoidf_(g2);
  int nm = nmask[n];
#pragma unroll
  for (int l2 = 0; l2 < 4; ++l2) {
    float v = nm ? u[l2] * g2 : side[((size_t)n * 4 + l2) * 32 + lane];
    out[((size_t)n * 4 + l2) * 32 + lane] = v;
  }
}

extern "C" void kernel_launch(void* const* d_in, const int* in_sizes, int n_in,
                              void* d_out, int out_size, void* d_ws, size_t ws_size,
                              hipStream_t stream) {
  const float* rot   = (const float*)d_in[0];
  const float* trans = (const float*)d_in[1];
  const float* rigid = (const float*)d_in[2];
  const float* side  = (const float*)d_in[3];
  const int*   batch = (const int*)d_in[4];
  const int*   xmask = (const int*)d_in[5];
  const int*   nmask = (const int*)d_in[6];
  const float* We1 = (const float*)d_in[7];
  const float* We2 = (const float*)d_in[8];
  const float* Wa1 = (const float*)d_in[9];
  const float* Wa2 = (const float*)d_in[10];
  const float* Wv  = (const float*)d_in[11];
  const float* Wg  = (const float*)d_in[12];
  const float* Wo  = (const float*)d_in[13];
  const float* Wf  = (const float*)d_in[14];
  (void)n_in; (void)out_size;

  const int n = in_sizes[1] / 3;      // 8192
  const int K = KNN;                  // 30 (reference constant)
  const long E = (long)n * K;

  char* ws = (char*)d_ws;
  size_t off = 0;
  auto alloc = [&](size_t bytes) -> char* {
    off = (off + 255) & ~(size_t)255;
    char* p = ws + off;
    off += bytes;
    return p;
  };
  _Float16* Xh    = (_Float16*)alloc((size_t)n * 4 * KPAD * 2);
  float*    y     = (float*)alloc((size_t)n * 4 * 64 * 4);
  float*    p     = (float*)alloc((size_t)n * 128 * 4);
  float*    q     = (float*)alloc((size_t)n * 128 * 4);
  int*      srcI  = (int*)alloc((size_t)E * 4);
  int*      valE  = (int*)alloc((size_t)E * 4);
  float*    lgts  = (float*)alloc((size_t)E * 8 * 4);
  _Float16* gateB = (_Float16*)alloc((size_t)E * 64 * 2);
  _Float16* Wpack = (_Float16*)alloc(WPACK_HALVES * 2);   // Wa1c|We1|We2|Wg contiguous
  _Float16* Wvt   = (_Float16*)alloc(64 * KPAD * 2);
  _Float16* Wa1at = (_Float16*)alloc(128 * KPAD * 2);
  _Float16* Wa1bt = (_Float16*)alloc(128 * KPAD * 2);
  (void)ws_size;

  auto g1 = [](int total) { return dim3((unsigned)((total + 255) / 256)); };
  // 1) pack weights into f16 N-major, K padded to WMMA chunk
  pack_wt<<<g1(64 * KPAD), 256, 0, stream>>>(Wv, Wvt, 64, 0, 67, KPAD, 64);
  pack_wt<<<g1(128 * KPAD), 256, 0, stream>>>(Wa1, Wa1at, 128, 0, 67, KPAD, 128);
  pack_wt<<<g1(128 * KPAD), 256, 0, stream>>>(Wa1, Wa1bt, 128, 67, 67, KPAD, 128);
  pack_wt<<<g1(128 * 32), 256, 0, stream>>>(Wa1, Wpack + WOFF_A1C, 128, 134, 32, 32, 128);
  pack_wt<<<g1(32 * 32), 256, 0, stream>>>(We1, Wpack + WOFF_WE1, 32, 0, 32, 32, 32);
  pack_wt<<<g1(32 * 32), 256, 0, stream>>>(We2, Wpack + WOFF_WE2, 32, 0, 32, 32, 32);
  pack_wt<<<g1(64 * 32), 256, 0, stream>>>(Wg, Wpack + WOFF_WG, 64, 0, 32, 32, 64);
  // 2) node features
  node_feat<<<g1(n * 4 * KPAD), 256, 0, stream>>>(rot, rigid, side, xmask, nmask, Xh, n);
  // 3) KNN (one wave32 per node)
  knn_kernel<<<dim3((unsigned)n), 32, 0, stream>>>(trans, batch, srcI, valE, n, K);
  // 4) node-level WMMA GEMMs (D-orthogonality removes all per-edge rotations):
  //    y = X @ Wv   [(n*4) x 96] x [96 x 64]
  gemm16<<<dim3((unsigned)(n * 4 / 16), 4), 32, 0, stream>>>(Xh, KPAD, Wvt, KPAD, 3, y, 64);
  //    p = X0 @ Wa1[0:67], q = X0 @ Wa1[67:134]   (A rows strided over l=0 rows)
  gemm16<<<dim3((unsigned)(n / 16), 8), 32, 0, stream>>>(Xh, 4 * KPAD, Wa1at, KPAD, 3, p, 128);
  gemm16<<<dim3((unsigned)(n / 16), 8), 32, 0, stream>>>(Xh, 4 * KPAD, Wa1bt, KPAD, 3, q, 128);
  // 5) fused per-edge MLP + attention logits + gates (16 edges per wave, WMMA chain,
  //    TDM-staged weights in LDS)
  edge_kernel<<<dim3((unsigned)((E + 15) / 16)), 32, 0, stream>>>(
      trans, srcI, valE, Wpack, Wa2, p, q, lgts, gateB, K, E);
  // 6) per-node softmax + gather-combine + Wo + scalar-gated Wf + masked scatter
  combine_kernel<<<dim3((unsigned)n), 32, 0, stream>>>(lgts, gateB, srcI, y, Wo, Wf,
                                                       nmask, side, (float*)d_out);
}
